// DiffPool_20658792694319
// MI455X (gfx1250) — compile-verified, matrix operand order checked by reading, not run on hardware
//
#include <hip/hip_runtime.h>
#include <hip/hip_bf16.h>

#define Nn 16384
#define Ff 256
#define Ee 524288
#define Bb 32
#define Kk 64
#define BK (Bb * Kk)     /* 2048 */
#define NPG (Nn / Bb)    /* 512 nodes per graph */
#define LSTR 40          /* LDS row stride in halfs: 80B -> 16B-aligned chunks */

typedef __attribute__((ext_vector_type(16))) __bf16 v16bf;
typedef __attribute__((ext_vector_type(8)))  float  v8f;

__device__ __forceinline__ unsigned short f2bf(float f) {
  unsigned int u = __builtin_bit_cast(unsigned int, f);
  u += 0x7FFFu + ((u >> 16) & 1u);   // round-to-nearest-even
  return (unsigned short)(u >> 16);
}

union FragBF { v16bf v; uint4 q[2]; unsigned short u[16]; };
union Half8  { uint4 q; unsigned short u[8]; };

// A stored [M][K] row-major halfs, B stored transposed [N][K]; stride LSTR.
// Lane l (r = l&15, base = (l>>4)*8) holds K = base..base+7 and base+16..base+23,
// matching the 16-bit A/B VGPR layout (two aligned ds_load_b128 per fragment).
__device__ __forceinline__ v16bf load_frag16(const unsigned short* tile, int lane) {
  int r  = lane & 15;
  int kb = (lane >> 4) << 3;
  const unsigned short* p = tile + r * LSTR + kb;
  FragBF f;
  f.q[0] = *(const uint4*)(p);
  f.q[1] = *(const uint4*)(p + 16);
  return f.v;
}

__device__ __forceinline__ v8f wmma_bf16(v16bf a, v16bf b, v8f c) {
  return __builtin_amdgcn_wmma_f32_16x16x32_bf16(false, a, false, b, (short)0, c,
                                                 false, false);
}

// C/D layout: VGPR r -> row = r + 8*(lane>>4), col = lane&15.
// LD is compile-time so the 8 stores share one base address + immediate offsets.
template <int LD>
__device__ __forceinline__ void store_tile16(float* out, int row0, int col0,
                                             v8f acc, int lane) {
  float* p = out + (size_t)(row0 + ((lane >> 4) << 3)) * LD + col0 + (lane & 15);
#pragma unroll
  for (int r = 0; r < 8; ++r) p[(size_t)r * LD] = acc[r];
}

// ---------------------------------------------------------------- utilities
__global__ void k_zero(int* __restrict__ p, int n) {
  int i = blockIdx.x * 256 + threadIdx.x;
  if (i < n) p[i] = 0;
}

__global__ void k_hist_dst(const int* __restrict__ dst, int* __restrict__ cnt) {
  int e = blockIdx.x * 256 + threadIdx.x;
  atomicAdd(&cnt[dst[e]], 1);
}

// deg = 1 (self loop) + in-degree; dinv = rsqrt(deg)
__global__ void k_dinv(const int* __restrict__ cnt, float* __restrict__ dinv) {
  int n = blockIdx.x * 256 + threadIdx.x;
  dinv[n] = rsqrtf(1.0f + (float)cnt[n]);
}

template <int ITEMS>
__global__ void k_scan(const int* __restrict__ cnt, int* __restrict__ offs,
                       int* __restrict__ cursor) {
  __shared__ int part[1024];
  int t = threadIdx.x;
  int base = t * ITEMS;
  int local[ITEMS];
  int sum = 0;
#pragma unroll
  for (int i = 0; i < ITEMS; ++i) { local[i] = sum; sum += cnt[base + i]; }
  part[t] = sum;
  __syncthreads();
  for (int off = 1; off < 1024; off <<= 1) {
    int v = (t >= off) ? part[t - off] : 0;
    __syncthreads();
    part[t] += v;
    __syncthreads();
  }
  int pre = (t == 0) ? 0 : part[t - 1];
#pragma unroll
  for (int i = 0; i < ITEMS; ++i) {
    int e = pre + local[i];
    offs[base + i] = e;
    cursor[base + i] = e;
  }
}

__global__ void k_scatter_dst(const int* __restrict__ src, const int* __restrict__ dst,
                              const float* __restrict__ dinv, int* __restrict__ cur,
                              int* __restrict__ csr_src, float* __restrict__ csr_nrm) {
  int e = blockIdx.x * 256 + threadIdx.x;
  int s = src[e], d = dst[e];
  int pos = atomicAdd(&cur[d], 1);
  csr_src[pos] = s;
  csr_nrm[pos] = dinv[s] * dinv[d];
}

__global__ void k_hist_pair(const int* __restrict__ src, const int* __restrict__ dst,
                            const int* __restrict__ batch, int* __restrict__ cnt) {
  int e = blockIdx.x * 256 + threadIdx.x;
  atomicAdd(&cnt[batch[src[e]] * Bb + batch[dst[e]]], 1);
}

__global__ void k_scatter_pair(const int* __restrict__ src, const int* __restrict__ dst,
                               const int* __restrict__ batch, int* __restrict__ cur,
                               int* __restrict__ ss, int* __restrict__ sd) {
  int e = blockIdx.x * 256 + threadIdx.x;
  int s = src[e], d = dst[e];
  int pos = atomicAdd(&cur[batch[s] * Bb + batch[d]], 1);
  ss[pos] = s;
  sd[pos] = d;
}

// ------------------------------------------------------- GEMM (x @ W) bf16 WMMA
// C[M,NCOLS] = A[M,KDIM] * W[KDIM,NCOLS]; block computes a 64x64 tile; 8 waves,
// 2 accumulator tiles per wave.
template <int KDIM, int NCOLS>
__global__ void k_gemm_bf16(const float* __restrict__ A, const float* __restrict__ W,
                            float* __restrict__ C) {
  __shared__ unsigned short Ash[64 * LSTR];
  __shared__ unsigned short Bsh[64 * LSTR];
  int t = threadIdx.x, lane = t & 31, wave = t >> 5;
  int m0 = blockIdx.x * 64, n0 = blockIdx.y * 64;
  int mi = wave & 3, ni0 = (wave >> 2) << 1;
  v8f acc0 = {0, 0, 0, 0, 0, 0, 0, 0}, acc1 = acc0;
  int row = t >> 2, cb = (t & 3) * 8;  // each thread stages 8 contiguous halfs
  for (int k0 = 0; k0 < KDIM; k0 += 32) {
    // A tile: Ash[m][k] = bf16(A[m0+m][k0+k])
    const float* ap = A + (size_t)(m0 + row) * KDIM + k0 + cb;
    float4 f0 = *(const float4*)(ap);
    float4 f1 = *(const float4*)(ap + 4);
    Half8 ha;
    ha.u[0] = f2bf(f0.x); ha.u[1] = f2bf(f0.y); ha.u[2] = f2bf(f0.z); ha.u[3] = f2bf(f0.w);
    ha.u[4] = f2bf(f1.x); ha.u[5] = f2bf(f1.y); ha.u[6] = f2bf(f1.z); ha.u[7] = f2bf(f1.w);
    *(uint4*)&Ash[row * LSTR + cb] = ha.q;
    // B tile transposed: Bsh[n][k] = bf16(W[k0+k][n0+n])
    Half8 hb;
#pragma unroll
    for (int i = 0; i < 8; ++i)
      hb.u[i] = f2bf(W[(size_t)(k0 + cb + i) * NCOLS + n0 + row]);
    *(uint4*)&Bsh[row * LSTR + cb] = hb.q;
    __syncthreads();
    v16bf a  = load_frag16(Ash + 16 * mi * LSTR, lane);
    v16bf b0 = load_frag16(Bsh + 16 * ni0 * LSTR, lane);
    v16bf b1 = load_frag16(Bsh + 16 * (ni0 + 1) * LSTR, lane);
    acc0 = wmma_bf16(a, b0, acc0);
    acc1 = wmma_bf16(a, b1, acc1);
    __syncthreads();
  }
  store_tile16<NCOLS>(C, m0 + 16 * mi, n0 + 16 * ni0, acc0, lane);
  store_tile16<NCOLS>(C, m0 + 16 * mi, n0 + 16 * (ni0 + 1), acc1, lane);
}

// -------------------------------------------- GCN aggregation (gather, CSR by dst)
// z is written directly as bf16 (only consumed by the WMMA xnext kernel).
__global__ void k_gather(const float* __restrict__ xw_e, const float* __restrict__ xw_a,
                         const float* __restrict__ dinv, const int* __restrict__ offs,
                         const int* __restrict__ cnt, const int* __restrict__ csr_src,
                         const float* __restrict__ csr_nrm, const float* __restrict__ be,
                         const float* __restrict__ ba, unsigned short* __restrict__ zbf,
                         float* __restrict__ slog) {
  int wave = threadIdx.x >> 5, lane = threadIdx.x & 31;
  int n = blockIdx.x * 8 + wave;
  float di = dinv[n], self = di * di;
  float accz[8], accs[2];
#pragma unroll
  for (int i = 0; i < 8; ++i) accz[i] = self * xw_e[(size_t)n * Ff + i * 32 + lane];
  accs[0] = self * xw_a[(size_t)n * Kk + lane];
  accs[1] = self * xw_a[(size_t)n * Kk + 32 + lane];
  int beg = offs[n], ne = cnt[n];
  for (int e = 0; e < ne; ++e) {
    int s = csr_src[beg + e];
    float nr = csr_nrm[beg + e];
    if (e + 1 < ne)
      __builtin_prefetch(xw_e + (size_t)csr_src[beg + e + 1] * Ff, 0, 1);
    const float* xr = xw_e + (size_t)s * Ff;
#pragma unroll
    for (int i = 0; i < 8; ++i) accz[i] += nr * xr[i * 32 + lane];
    const float* xa = xw_a + (size_t)s * Kk;
    accs[0] += nr * xa[lane];
    accs[1] += nr * xa[32 + lane];
  }
#pragma unroll
  for (int i = 0; i < 8; ++i)
    zbf[(size_t)n * Ff + i * 32 + lane] = f2bf(accz[i] + be[i * 32 + lane]);
  slog[(size_t)n * Kk + lane] = accs[0] + ba[lane];
  slog[(size_t)n * Kk + 32 + lane] = accs[1] + ba[32 + lane];
}

// ------------------------------------------------------------ softmax -> bf16
__global__ void k_softmax(const float* __restrict__ slog, unsigned short* __restrict__ slh) {
  int wave = threadIdx.x >> 5, lane = threadIdx.x & 31;
  int n = blockIdx.x * 8 + wave;
  float v0 = slog[(size_t)n * Kk + lane];
  float v1 = slog[(size_t)n * Kk + 32 + lane];
  float m = fmaxf(v0, v1);
  for (int o = 16; o > 0; o >>= 1) m = fmaxf(m, __shfl_xor(m, o, 32));
  float e0 = __expf(v0 - m), e1 = __expf(v1 - m);
  float s = e0 + e1;
  for (int o = 16; o > 0; o >>= 1) s += __shfl_xor(s, o, 32);
  float inv = 1.0f / s;
  slh[(size_t)n * Kk + lane] = f2bf(e0 * inv);
  slh[(size_t)n * Kk + 32 + lane] = f2bf(e1 * inv);
}

// --------------- anext block (b1,b2): sum over bucket edges of s[s]⊗s[d], WMMA
__global__ void k_anext(const unsigned short* __restrict__ slh,
                        const int* __restrict__ ss, const int* __restrict__ sd,
                        const int* __restrict__ offs, const int* __restrict__ cnt,
                        float* __restrict__ anext) {
  __shared__ unsigned short Ash[64 * LSTR];
  __shared__ unsigned short Bsh[64 * LSTR];
  int p = blockIdx.x, b1 = p >> 5, b2 = p & 31;
  int t = threadIdx.x, lane = t & 31, wave = t >> 5;
  int mi = wave & 3, ni0 = (wave >> 2) << 1;
  v8f acc0 = {0, 0, 0, 0, 0, 0, 0, 0}, acc1 = acc0;
  int beg = offs[p], ne = cnt[p];
  int nch = (ne + 31) >> 5;
  int j = t & 31, seg = t >> 5;   // thread stages rows seg*8..+7 of edge column j
  for (int c = 0; c < nch; ++c) {
    int ei = c * 32 + j;
    Half8 ua, ub;
    if (ei < ne) {
      int s = ss[beg + ei], d = sd[beg + ei];
      ua.q = *(const uint4*)(slh + (size_t)s * Kk + seg * 8);
      ub.q = *(const uint4*)(slh + (size_t)d * Kk + seg * 8);
    } else {
      ua.q = make_uint4(0u, 0u, 0u, 0u);
      ub.q = ua.q;
    }
#pragma unroll
    for (int i = 0; i < 8; ++i) {
      Ash[(seg * 8 + i) * LSTR + j] = ua.u[i];   // A = Ssrc^T  [c1][edge]
      Bsh[(seg * 8 + i) * LSTR + j] = ub.u[i];   // B^T = Sdst^T [c2][edge]
    }
    __syncthreads();
    v16bf a   = load_frag16(Ash + 16 * mi * LSTR, lane);
    v16bf bb0 = load_frag16(Bsh + 16 * ni0 * LSTR, lane);
    v16bf bb1 = load_frag16(Bsh + 16 * (ni0 + 1) * LSTR, lane);
    acc0 = wmma_bf16(a, bb0, acc0);
    acc1 = wmma_bf16(a, bb1, acc1);
    __syncthreads();
  }
  store_tile16<BK>(anext, b1 * 64 + 16 * mi, b2 * 64 + 16 * ni0, acc0, lane);
  store_tile16<BK>(anext, b1 * 64 + 16 * mi, b2 * 64 + 16 * (ni0 + 1), acc1, lane);
}

// --------------------- xnext: per graph g, S_g^T [64x512] @ Z_g [512x256] (WMMA)
__global__ void k_xnext(const unsigned short* __restrict__ slh,
                        const unsigned short* __restrict__ zbf,
                        float* __restrict__ xnext) {
  __shared__ unsigned short Ash[64 * LSTR];
  __shared__ unsigned short Bsh[64 * LSTR];
  int g = blockIdx.x, fc = blockIdx.y;
  int t = threadIdx.x, lane = t & 31, wave = t >> 5;
  int mi = wave & 3, ni0 = (wave >> 2) << 1;
  v8f acc0 = {0, 0, 0, 0, 0, 0, 0, 0}, acc1 = acc0;
  int j = t & 31, seg = t >> 5;
  for (int c = 0; c < NPG / 32; ++c) {
    int node = g * NPG + c * 32 + j;
    Half8 ua, ub;
    ua.q = *(const uint4*)(slh + (size_t)node * Kk + seg * 8);
    ub.q = *(const uint4*)(zbf + (size_t)node * Ff + fc * 64 + seg * 8);
#pragma unroll
    for (int i = 0; i < 8; ++i) {
      Ash[(seg * 8 + i) * LSTR + j] = ua.u[i];   // A = S^T [c1][node]
      Bsh[(seg * 8 + i) * LSTR + j] = ub.u[i];   // B^T = Z^T [f][node]
    }
    __syncthreads();
    v16bf a   = load_frag16(Ash + 16 * mi * LSTR, lane);
    v16bf bb0 = load_frag16(Bsh + 16 * ni0 * LSTR, lane);
    v16bf bb1 = load_frag16(Bsh + 16 * (ni0 + 1) * LSTR, lane);
    acc0 = wmma_bf16(a, bb0, acc0);
    acc1 = wmma_bf16(a, bb1, acc1);
    __syncthreads();
  }
  store_tile16<Ff>(xnext, g * 64 + 16 * mi, fc * 64 + 16 * ni0, acc0, lane);
  store_tile16<Ff>(xnext, g * 64 + 16 * mi, fc * 64 + 16 * (ni0 + 1), acc1, lane);
}

__global__ void k_batchnext(int* __restrict__ bn) {
  int i = blockIdx.x * 256 + threadIdx.x;
  bn[i] = i >> 6;  // i / K
}

// ---------------------------------------------------------------------------
extern "C" void kernel_launch(void* const* d_in, const int* in_sizes, int n_in,
                              void* d_out, int out_size, void* d_ws, size_t ws_size,
                              hipStream_t stream) {
  const float* x     = (const float*)d_in[0];
  const int*   eidx  = (const int*)d_in[1];
  const int*   batch = (const int*)d_in[2];
  const float* W_e   = (const float*)d_in[3];
  const float* b_e   = (const float*)d_in[4];
  const float* W_a   = (const float*)d_in[5];
  const float* b_a   = (const float*)d_in[6];
  const int* src = eidx;
  const int* dst = eidx + Ee;

  float* out   = (float*)d_out;
  float* xnext = out;                              // [2048, 256]
  float* anext = out + (size_t)BK * Ff;            // [2048, 2048]
  int*   bnext = (int*)(out + (size_t)BK * Ff + (size_t)BK * BK);  // [2048]

  char* w = (char*)d_ws;
  float* dinv    = (float*)w;           w += (size_t)Nn * 4;
  int*   cnt_d   = (int*)w;             w += (size_t)Nn * 4;
  int*   offs_d  = (int*)w;             w += (size_t)Nn * 4;
  int*   cur_d   = (int*)w;             w += (size_t)Nn * 4;
  int*   cnt_p   = (int*)w;             w += 4096;
  int*   offs_p  = (int*)w;             w += 4096;
  int*   cur_p   = (int*)w;             w += 4096;
  int*   csr_src = (int*)w;             w += (size_t)Ee * 4;
  float* csr_nrm = (float*)w;           w += (size_t)Ee * 4;
  int*   ss      = (int*)w;             w += (size_t)Ee * 4;
  int*   sd      = (int*)w;             w += (size_t)Ee * 4;
  float* xw_e    = (float*)w;           w += (size_t)Nn * Ff * 4;
  float* xw_a    = (float*)w;           w += (size_t)Nn * Kk * 4;
  float* slog    = (float*)w;           w += (size_t)Nn * Kk * 4;
  unsigned short* zbf = (unsigned short*)w;  w += (size_t)Nn * Ff * 2;  // [N,F] bf16
  unsigned short* slh = (unsigned short*)w;  // [N, K] bf16

  // 1) degree histogram + dinv + CSR-by-dst (counting sort)
  k_zero<<<Nn / 256, 256, 0, stream>>>(cnt_d, Nn);
  k_zero<<<4, 256, 0, stream>>>(cnt_p, 1024);
  k_hist_dst<<<Ee / 256, 256, 0, stream>>>(dst, cnt_d);
  k_dinv<<<Nn / 256, 256, 0, stream>>>(cnt_d, dinv);
  k_scan<16><<<1, 1024, 0, stream>>>(cnt_d, offs_d, cur_d);
  k_scatter_dst<<<Ee / 256, 256, 0, stream>>>(src, dst, dinv, cur_d, csr_src, csr_nrm);

  // 2) feature GEMMs (bf16 WMMA, f32 accumulate)
  k_gemm_bf16<Ff, Ff><<<dim3(Nn / 64, Ff / 64), 256, 0, stream>>>(x, W_e, xw_e);
  k_gemm_bf16<Ff, Kk><<<dim3(Nn / 64, Kk / 64), 256, 0, stream>>>(x, W_a, xw_a);

  // 3) GCN aggregation (gather) + softmax
  k_gather<<<Nn / 8, 256, 0, stream>>>(xw_e, xw_a, dinv, offs_d, cnt_d, csr_src,
                                       csr_nrm, b_e, b_a, zbf, slog);
  k_softmax<<<Nn / 8, 256, 0, stream>>>(slog, slh);

  // 4) bucket edges by (batch[src], batch[dst]) pair
  k_hist_pair<<<Ee / 256, 256, 0, stream>>>(src, dst, batch, cnt_p);
  k_scan<1><<<1, 1024, 0, stream>>>(cnt_p, offs_p, cur_p);
  k_scatter_pair<<<Ee / 256, 256, 0, stream>>>(src, dst, batch, cur_p, ss, sd);

  // 5) pooled adjacency and pooled features (WMMA)
  k_anext<<<Bb * Bb, 256, 0, stream>>>(slh, ss, sd, offs_p, cnt_p, anext);
  k_xnext<<<dim3(Bb, Ff / 64), 256, 0, stream>>>(slh, zbf, xnext);
  k_batchnext<<<BK / 256, 256, 0, stream>>>(bnext);
}
